// WaveNet_21036749816518
// MI455X (gfx1250) — matrix operand order repaired and checked by reading, not compiled, not run on hardware
//
#include <hip/hip_runtime.h>
#include <hip/hip_bf16.h>

typedef __bf16 bf16_t;
typedef bf16_t v16bf __attribute__((ext_vector_type(16)));
typedef bf16_t v8bf  __attribute__((ext_vector_type(8)));
typedef float  v8f   __attribute__((ext_vector_type(8)));
typedef unsigned int v4u __attribute__((ext_vector_type(4)));

constexpr int BB    = 32;    // batch
constexpr int TT    = 4096;  // time
constexpr int PAD   = 512;   // zero prefix per batch row == max dilation
constexpr int TP    = TT + PAD;  // padded time stride (4608)
constexpr int CRES  = 64;
constexpr int CSKIP = 256;
constexpr int LTOT  = 20;
constexpr int TPB   = 256;   // 8 wave32 per block
constexpr int LROW  = 72;    // LDS row stride (channels) -> bank-conflict padding

static __device__ __forceinline__ v8f wmma_bf16(v16bf a, v16bf b, v8f c) {
  // D = A(16x32) * B(32x16) + C, f32 accumulate
  return __builtin_amdgcn_wmma_f32_16x16x32_bf16(false, a, false, b, (short)0, c,
                                                 false, false);
}

static __device__ __forceinline__ float tanh_fast(float x) {
#if __has_builtin(__builtin_amdgcn_tanhf)
  return __builtin_amdgcn_tanhf(x);   // v_tanh_f32 (gfx1250 trans op)
#else
  return tanhf(x);
#endif
}
static __device__ __forceinline__ float sigmoid_fast(float x) {
  // sigma(x) = 0.5*tanh(x/2) + 0.5  -> 1 trans + 2 VALU
  return __builtin_fmaf(0.5f, tanh_fast(0.5f * x), 0.5f);
}

// ---------------------------------------------------------------------------
// Repack fp32 conv weights [20][mtiles*16][64][ktaps] into bf16 WMMA A-tiles
// [20][mtiles][ksteps][32 lanes][16 halves] in contiguous K order, shared with the B loads:
//   lane l (row M = l&15), half j -> K = (l>>4)*16 + j
// K-step order for ktaps==2: s0,s1 = tap0 (x[t-d]) ch 0-31 / 32-63,
//                            s2,s3 = tap1 (x[t])   ch 0-31 / 32-63.
// ---------------------------------------------------------------------------
__global__ __launch_bounds__(TPB) void repack_weights(
    const float* __restrict__ w, bf16_t* __restrict__ dst,
    int mtiles, int ksteps, int ktaps) {
  const int tid  = blockIdx.x * TPB + threadIdx.x;
  const int lane = tid & 31;
  int rr = tid >> 5;
  const int s  = rr % ksteps; rr /= ksteps;
  const int mt = rr % mtiles;
  const int l  = rr / mtiles;
  if (l >= LTOT) return;
  const int m    = mt * 16 + (lane & 15);
  const int tap  = (ktaps == 2) ? (s >> 1) : 0;
  const int cgrp = (ktaps == 2) ? (s & 1) : s;
  bf16_t* o = dst + (size_t)((((l * mtiles + mt) * ksteps + s) * 32) + lane) * 16;
  for (int j = 0; j < 16; ++j) {
    const int klocal = (lane >> 4) * 16 + j;
    const int cin    = cgrp * 32 + klocal;
    o[j] = (bf16_t)w[((size_t)(l * mtiles * 16 + m) * 64 + cin) * ktaps + tap];
  }
}

// ---------------------------------------------------------------------------
// Zero the 512-timestep causal pads of both h ping-pong buffers.
// ---------------------------------------------------------------------------
__global__ __launch_bounds__(TPB) void zero_pads(bf16_t* __restrict__ hA,
                                                 bf16_t* __restrict__ hB) {
  const int tid = blockIdx.x * TPB + threadIdx.x;  // B*PAD*8 threads
  const int g  = tid & 7;
  const int tr = (tid >> 3) & (PAD - 1);
  const int b  = tid >> 12;                        // PAD*8 == 4096
  const size_t off = ((size_t)b * TP + tr) * CRES + g * 8;
  v8bf z = {};
  *(v8bf*)(hA + off) = z;
  *(v8bf*)(hB + off) = z;
}

// ---------------------------------------------------------------------------
// Input 1x1 conv: h[b][PAD+t][ch] = W_in(64x8) @ x[b][:][t] + b_in  (K=8: VALU)
// ---------------------------------------------------------------------------
__global__ __launch_bounds__(TPB) void input_conv(
    const float* __restrict__ x, const float* __restrict__ w_in,
    const float* __restrict__ b_in, bf16_t* __restrict__ h) {
  const int tid = blockIdx.x * TPB + threadIdx.x;  // B*T*8 threads
  const int g = tid & 7;
  const int t = (tid >> 3) & (TT - 1);
  const int b = tid >> 15;
  float xv[8];
  for (int c = 0; c < 8; ++c) xv[c] = x[((size_t)b * 8 + c) * TT + t];
  v8bf o;
  for (int r = 0; r < 8; ++r) {
    const int ch = g * 8 + r;
    float a = b_in[ch];
    for (int c = 0; c < 8; ++c) a += w_in[ch * 8 + c] * xv[c];
    o[r] = (bf16_t)a;
  }
  *(v8bf*)(h + ((size_t)b * TP + PAD + t) * CRES + g * 8) = o;
}

// ---------------------------------------------------------------------------
// One gated residual layer, fully fused. One wave per 16-timestep tile:
//   f/g conv (K=128, 32 WMMA), gate in f32 via v_tanh_f32,
//   z -> LDS (bf16, [t][ch]), z -> global (NT), residual GEMM (8 WMMA).
// Causal pad handled by the zeroed 512-step prefix: no EXEC divergence.
// ---------------------------------------------------------------------------
__global__ __launch_bounds__(TPB) void wavenet_layer(
    const bf16_t* __restrict__ hin, bf16_t* __restrict__ hout,
    bf16_t* __restrict__ zout,
    const bf16_t* __restrict__ wfp, const bf16_t* __restrict__ wgp,
    const bf16_t* __restrict__ wrp,
    const float* __restrict__ bf_, const float* __restrict__ bg_,
    const float* __restrict__ br_, int dil) {
  __shared__ __align__(32) bf16_t zsh[(TPB / 32) * 16 * LROW];
  const int lane = threadIdx.x & 31;
  const int wid  = threadIdx.x >> 5;
  const int gw   = blockIdx.x * (TPB / 32) + wid;  // 0 .. B*T/16-1
  const int b    = gw >> 8;                        // T/16 == 256
  const int t0   = (gw & 255) << 4;
  const int col  = lane & 15;                      // timestep column / N
  const int kh   = lane >> 4;                      // K-half select
  const int t    = t0 + col;
  const size_t rowB = (size_t)b * TP + PAD;        // points at t==0

  // ---- gated dilated conv: [f;g] = W(64x128) @ [h(t-d); h(t)] ----
  v8f facc[4] = {}, gacc[4] = {};
  for (int s = 0; s < 4; ++s) {
    const int ts = t - ((s < 2) ? dil : 0);        // tap0 -> t-d, tap1 -> t
    const int c0 = (s & 1) * 32 + kh * 16;
    v16bf bt = *(const v16bf*)(hin + (rowB + ts) * CRES + c0);  // pad is zeros
    for (int m = 0; m < 4; ++m) {
      v16bf af = *(const v16bf*)(wfp + (size_t)((m * 4 + s) * 32 + lane) * 16);
      facc[m] = wmma_bf16(af, bt, facc[m]);
      v16bf ag = *(const v16bf*)(wgp + (size_t)((m * 4 + s) * 32 + lane) * 16);
      gacc[m] = wmma_bf16(ag, bt, gacc[m]);
    }
  }

  // ---- bias + tanh*sigmoid -> z staged in LDS as [t][ch] (padded rows) ----
  bf16_t* myz = zsh + wid * (16 * LROW);
  for (int m = 0; m < 4; ++m) {
    const int ch0 = m * 16 + kh * 8;               // C-layout: M = r + kh*8
    v8bf zv;
    for (int r = 0; r < 8; ++r) {
      float fv = tanh_fast(facc[m][r] + bf_[ch0 + r]);
      float gv = sigmoid_fast(gacc[m][r] + bg_[ch0 + r]);
      zv[r] = (bf16_t)(fv * gv);
    }
    *(v8bf*)(myz + col * LROW + ch0) = zv;
  }

  // ---- stream z tile to global [B][T][64] (non-temporal: read-once) ----
  {
    bf16_t* zg = zout + ((size_t)b * TT + t0) * CRES;
    for (int i = 0; i < 4; ++i) {
      const int e  = i * 256 + lane * 8;           // 16ts*64ch elements
      const int ts = e >> 6;
      const int ch = e & 63;
      __builtin_nontemporal_store(*(const v4u*)(myz + ts * LROW + ch),
                                  (v4u*)(zg + e));
    }
  }

  // ---- residual: h' = h + W_r(64x64) @ z + b_r ----
  v8f racc[4];
  for (int m = 0; m < 4; ++m) {
    const int ch0 = m * 16 + kh * 8;
    v8bf hold = *(const v8bf*)(hin + (rowB + t) * CRES + ch0);
    for (int r = 0; r < 8; ++r) racc[m][r] = (float)hold[r] + br_[ch0 + r];
  }
  for (int s = 0; s < 2; ++s) {
    v16bf zb = *(const v16bf*)(myz + col * LROW + s * 32 + kh * 16);
    for (int m = 0; m < 4; ++m) {
      v16bf ar = *(const v16bf*)(wrp + (size_t)((m * 2 + s) * 32 + lane) * 16);
      racc[m] = wmma_bf16(ar, zb, racc[m]);
    }
  }
  for (int m = 0; m < 4; ++m) {
    const int ch0 = m * 16 + kh * 8;
    v8bf o;
    for (int r = 0; r < 8; ++r) o[r] = (bf16_t)racc[m][r];
    *(v8bf*)(hout + (rowB + t) * CRES + ch0) = o;
  }
}

// ---------------------------------------------------------------------------
// Skip GEMM: skips[b,:,t] += W_s(256x64) @ z_l over layers [lb,le).
// Wave = 16 timesteps x 64 out channels (quad of 4 mtiles); K = (le-lb)*64.
// pathA: single call, K=1280, init with summed bias. pathB: per-layer RMW.
// ---------------------------------------------------------------------------
__global__ __launch_bounds__(TPB) void skip_gemm(
    const bf16_t* __restrict__ zbuf,   // [(le-lb)][B][T][64]
    const bf16_t* __restrict__ wsp,    // [20][16][2][32][16]
    const float* __restrict__ bs_,     // [20][256]
    float* __restrict__ out,           // [B][256][T]
    int lb, int le, int accum) {
  const int lane = threadIdx.x & 31;
  const int wid  = threadIdx.x >> 5;
  const int gw   = blockIdx.x * (TPB / 32) + wid;  // B*(T/16)*4 waves
  const int quad = gw & 3;
  const int tile = gw >> 2;
  const int b    = tile >> 8;
  const int t0   = (tile & 255) << 4;
  const int col  = lane & 15;
  const int kh   = lane >> 4;

  v8f acc[4] = {};
  for (int l = lb; l < le; ++l) {
    const size_t zrow = (((size_t)(l - lb) * BB + b) * TT + t0 + col) * CRES;
    for (int s = 0; s < 2; ++s) {
      // z is a one-pass stream: non-temporal load (2x b128)
      v4u lo = __builtin_nontemporal_load(
          (const v4u*)(zbuf + zrow + s * 32 + kh * 16));
      v4u hi = __builtin_nontemporal_load(
          (const v4u*)(zbuf + zrow + s * 32 + kh * 16 + 8));
      struct { v4u a, b; } pair{lo, hi};
      v16bf zb;
      __builtin_memcpy(&zb, &pair, sizeof(zb));
      for (int m = 0; m < 4; ++m) {
        const int mt = quad * 4 + m;
        v16bf aw = *(const v16bf*)(wsp +
            (size_t)(((l * 16 + mt) * 2 + s) * 32 + lane) * 16);
        acc[m] = wmma_bf16(aw, zb, acc[m]);
      }
    }
  }
  for (int m = 0; m < 4; ++m) {
    const int ch0 = quad * 64 + m * 16 + kh * 8;
    float bsum[8] = {};
    for (int l = lb; l < le; ++l)
      for (int r = 0; r < 8; ++r) bsum[r] += bs_[l * CSKIP + ch0 + r];
    for (int r = 0; r < 8; ++r) {
      const size_t idx = ((size_t)b * CSKIP + ch0 + r) * TT + t0 + col;
      float v = acc[m][r] + bsum[r];
      if (accum) v += __builtin_nontemporal_load(out + idx);
      __builtin_nontemporal_store(v, out + idx);
    }
  }
}

// ---------------------------------------------------------------------------
extern "C" void kernel_launch(void* const* d_in, const int* in_sizes, int n_in,
                              void* d_out, int out_size, void* d_ws,
                              size_t ws_size, hipStream_t stream) {
  (void)in_sizes; (void)n_in; (void)out_size;
  const float* x    = (const float*)d_in[0];
  const float* w_in = (const float*)d_in[1];
  const float* b_in = (const float*)d_in[2];
  const float* w_f  = (const float*)d_in[3];
  const float* b_f  = (const float*)d_in[4];
  const float* w_g  = (const float*)d_in[5];
  const float* b_g  = (const float*)d_in[6];
  const float* w_s  = (const float*)d_in[7];
  const float* b_s  = (const float*)d_in[8];
  const float* w_r  = (const float*)d_in[9];
  const float* b_r  = (const float*)d_in[10];
  float* out = (float*)d_out;

  const size_t HT   = (size_t)BB * TP * CRES;  // padded h elems (9,437,184)
  const size_t BT64 = (size_t)BB * TT * CRES;  // z slice elems  (8,388,608)
  char* p = (char*)d_ws;
  bf16_t* hA   = (bf16_t*)p; p += HT * 2;                   // 18 MB
  bf16_t* hB   = (bf16_t*)p; p += HT * 2;                   // 18 MB
  bf16_t* wf_p = (bf16_t*)p; p += (size_t)LTOT * 16 * 512 * 2;
  bf16_t* wg_p = (bf16_t*)p; p += (size_t)LTOT * 16 * 512 * 2;
  bf16_t* wr_p = (bf16_t*)p; p += (size_t)LTOT * 8 * 512 * 2;
  bf16_t* ws_p = (bf16_t*)p; p += (size_t)LTOT * 32 * 512 * 2;
  bf16_t* zbuf = (bf16_t*)p;
  const size_t used = (size_t)(p - (char*)d_ws);
  const bool pathA = ws_size >= used + (size_t)LTOT * BT64 * 2;  // 320 MB z

  // Weight repack (tiny; deterministic every call)
  repack_weights<<<40, TPB, 0, stream>>>(w_f, wf_p, 4, 4, 2);
  repack_weights<<<40, TPB, 0, stream>>>(w_g, wg_p, 4, 4, 2);
  repack_weights<<<20, TPB, 0, stream>>>(w_r, wr_p, 4, 2, 1);
  repack_weights<<<80, TPB, 0, stream>>>(w_s, ws_p, 16, 2, 1);

  zero_pads<<<(BB * PAD * 8) / TPB, TPB, 0, stream>>>(hA, hB);
  input_conv<<<(BB * TT * 8) / TPB, TPB, 0, stream>>>(x, w_in, b_in, hA);

  bf16_t* hc = hA;
  bf16_t* hn = hB;
  const int layer_blocks = BB * (TT / 16) / (TPB / 32);       // 1024
  const int skip_blocks  = BB * (TT / 16) * 4 / (TPB / 32);   // 4096
  for (int i = 0; i < LTOT; ++i) {
    const int d = 1 << (i % 10);
    bf16_t* zdst = pathA ? (zbuf + (size_t)i * BT64) : zbuf;
    wavenet_layer<<<layer_blocks, TPB, 0, stream>>>(
        hc, hn, zdst,
        wf_p + (size_t)i * 16 * 512, wg_p + (size_t)i * 16 * 512,
        wr_p + (size_t)i * 8 * 512,
        b_f + i * CRES, b_g + i * CRES, b_r + i * CRES, d);
    if (!pathA)
      skip_gemm<<<skip_blocks, TPB, 0, stream>>>(zbuf, ws_p, b_s, out,
                                                 i, i + 1, i > 0);
    bf16_t* tmp = hc; hc = hn; hn = tmp;
  }
  if (pathA)
    skip_gemm<<<skip_blocks, TPB, 0, stream>>>(zbuf, ws_p, b_s, out,
                                               0, LTOT, 0);
}
